// ConvAttention_36593121362519
// MI455X (gfx1250) — matrix operand order repaired and checked by reading, not compiled, not run on hardware
//
#include <hip/hip_runtime.h>
#include <math.h>

// ---------------------------------------------------------------------------
// ConvAttention for MI455X (gfx1250, wave32, WMMA 16x16x32 bf16)
// B=8, C=256, E=512, NH=8, HD=64, N=H*W=1024
//
// Fragments load as 2x ds_load_b128; all 4 fragments of a k-step are loaded
// into distinct registers before the 4 WMMAs so DS latency pipelines.
// Attention computes S^T = K*Q^T and O^T = V*P^T so softmax is lane-local
// (one xor-16 shuffle per reduction). P^T writes pack 8 consecutive keys into
// one ds_store_b128. All epilogues bounce through LDS for coalesced b128
// global stores. Layouts: xt (B,N,C); Q,K,AO (B,N,E); V (B,E,N).
// ---------------------------------------------------------------------------

typedef __attribute__((ext_vector_type(16))) __bf16 v16bf;
typedef __attribute__((ext_vector_type(8)))  float  v8f;

#define BDIM  8
#define CDIM  256
#define EDIM  512
#define NHEAD 8
#define HDIM  64
#define NPIX  1024

__device__ __forceinline__ v16bf load_frag_a(const __bf16* row, int half) {
  // element i -> K = i + 16*half
  union { uint4 q[2]; v16bf v; } u;
  u.q[0] = *(const uint4*)(row + 16 * half);
  u.q[1] = *(const uint4*)(row + 16 * half + 8);
  return u.v;
}
__device__ __forceinline__ v16bf load_frag_b(const __bf16* p) {
  // element i -> K = i + 16*half: caller passes col_row + 16*half
  union { uint4 q[2]; v16bf v; } u;
  u.q[0] = *(const uint4*)(p);
  u.q[1] = *(const uint4*)(p + 8);
  return u.v;
}
// Pack 8 f32 -> 8 bf16, single b128 store (dst 16B aligned).
__device__ __forceinline__ void store_pack8_bf16(__bf16* dst, v8f v) {
  union { __bf16 e[8]; uint4 q; } u;
#pragma unroll
  for (int r = 0; r < 8; ++r) u.e[r] = (__bf16)v[r];
  *(uint4*)dst = u.q;
}

// ---------------------------------------------------------------------------
__global__ void f32_to_bf16_kernel(const float* __restrict__ in,
                                   __bf16* __restrict__ out, int n, float scale) {
  int i = blockIdx.x * blockDim.x + threadIdx.x;
  if (i < n) out[i] = (__bf16)(in[i] * scale);
}

// x (B,C,N) fp32 -> xt (B,N,C) bf16, LDS-tiled transpose.
__global__ __launch_bounds__(256)
void transpose_cvt_kernel(const float* __restrict__ in, __bf16* __restrict__ out) {
  __shared__ float tile[32][33];
  const int n0 = blockIdx.x * 32, c0 = blockIdx.y * 32, b = blockIdx.z;
  for (int i = threadIdx.y; i < 32; i += 8)
    tile[i][threadIdx.x] = in[((size_t)b * CDIM + c0 + i) * NPIX + n0 + threadIdx.x];
  __syncthreads();
  for (int i = threadIdx.y; i < 32; i += 8)
    out[((size_t)b * NPIX + n0 + i) * CDIM + c0 + threadIdx.x] =
        (__bf16)tile[threadIdx.x][i];
}

// ---------------------------------------------------------------------------
// GEMM + bias: out[p][e] = sum_k X[b][p][k] * W[e][k] + bias[e]*bscale
// M = 64 pixels (blockIdx.x), N = 64 features (blockIdx.y), 4 waves.
// OUT_LAYOUT: 0 = (B,N,E) bf16, 1 = (B,E,N) bf16, 2 = (B,E,N) f32
// ---------------------------------------------------------------------------
template <int OUT_LAYOUT>
__global__ __launch_bounds__(128)
void gemm_bias_kernel(const __bf16* __restrict__ X,   // (B, NPIX, Kdim)
                      const __bf16* __restrict__ W,   // (EDIM, Kdim) row-major
                      const float*  __restrict__ bias,
                      void* __restrict__ outv, int Kdim, float bscale) {
  __shared__ __align__(16) char smem[64 * 68 * 4];   // 17408 B >= 2*64*40*2
  __bf16 (*As)[40] = (__bf16 (*)[40])smem;
  __bf16 (*Bs)[40] = (__bf16 (*)[40])(smem + 64 * 40 * 2);
  __bf16 (*Ob)[72] = (__bf16 (*)[72])smem;           // bf16 out stage
  float  (*Of)[68] = (float  (*)[68])smem;           // f32  out stage

  const int t    = threadIdx.x;
  const int lane = t & 31;
  const int wave = t >> 5;
  const int half = lane >> 4;
  const int l16  = lane & 15;

  const int b  = blockIdx.z;
  const int p0 = blockIdx.x * 64;
  const int e0 = blockIdx.y * 64;

  const __bf16* Xb = X + ((size_t)b * NPIX + p0) * Kdim;
  const __bf16* Wb = W + (size_t)e0 * Kdim;

  v8f acc[4];
#pragma unroll
  for (int nt = 0; nt < 4; ++nt)
#pragma unroll
    for (int r = 0; r < 8; ++r) acc[nt][r] = 0.0f;

  for (int kk = 0; kk < Kdim; kk += 32) {
    __syncthreads();
#pragma unroll
    for (int i = 0; i < 2; ++i) {
      int j = t + i * 128;            // 256 uint4 per tile
      int row = j >> 2;
      int off = (j & 3) * 8;
      *(uint4*)(&As[row][off]) = *(const uint4*)(Xb + (size_t)row * Kdim + kk + off);
      *(uint4*)(&Bs[row][off]) = *(const uint4*)(Wb + (size_t)row * Kdim + kk + off);
      if (kk + 32 < Kdim) {           // global_prefetch_b8 next K tile
        __builtin_prefetch(Xb + (size_t)row * Kdim + kk + 32 + off, 0, 0);
        __builtin_prefetch(Wb + (size_t)row * Kdim + kk + 32 + off, 0, 0);
      }
    }
    __syncthreads();

    // Load A + all 4 B fragments first (distinct regs), then issue 4 WMMAs.
    v16bf a = load_frag_a(&As[wave * 16 + l16][0], half);
    v16bf bb[4];
#pragma unroll
    for (int nt = 0; nt < 4; ++nt)
      bb[nt] = load_frag_b(&Bs[nt * 16 + l16][16 * half]);
#pragma unroll
    for (int nt = 0; nt < 4; ++nt)
      acc[nt] = __builtin_amdgcn_wmma_f32_16x16x32_bf16(
          false, a, false, bb[nt], (short)0, acc[nt], false, false);
  }

  // Epilogue: stage 64x64 in LDS, stream out coalesced.
  // D element r -> M(pixel) = r + 8*half, N(feature) = lane&15.
  __syncthreads();
#pragma unroll
  for (int nt = 0; nt < 4; ++nt) {
    int el = nt * 16 + l16;
    float bv = bias[e0 + el] * bscale;
    v8f v;
#pragma unroll
    for (int r = 0; r < 8; ++r) v[r] = acc[nt][r] + bv;
    if (OUT_LAYOUT == 0) {
      // pixel-major stage: column writes, scalar (once per kernel).
#pragma unroll
      for (int r = 0; r < 8; ++r) Ob[wave * 16 + r + 8 * half][el] = (__bf16)v[r];
    } else if (OUT_LAYOUT == 1) {
      store_pack8_bf16(&Ob[el][wave * 16 + 8 * half], v);   // 8 consecutive pixels
    } else {
      union { float e[8]; uint4 q[2]; } u;
#pragma unroll
      for (int r = 0; r < 8; ++r) u.e[r] = v[r];
      *(uint4*)(&Of[el][wave * 16 + 8 * half])     = u.q[0];
      *(uint4*)(&Of[el][wave * 16 + 8 * half + 4]) = u.q[1];
    }
  }
  __syncthreads();
  if (OUT_LAYOUT == 0) {
    __bf16* o = (__bf16*)outv;
#pragma unroll
    for (int i = 0; i < 4; ++i) {
      int j = t + i * 128, row = j >> 3, off = (j & 7) * 8;
      *(uint4*)(o + ((size_t)b * NPIX + p0 + row) * EDIM + e0 + off) =
          *(const uint4*)(&Ob[row][off]);
    }
  } else if (OUT_LAYOUT == 1) {
    __bf16* o = (__bf16*)outv;
#pragma unroll
    for (int i = 0; i < 4; ++i) {
      int j = t + i * 128, row = j >> 3, off = (j & 7) * 8;
      *(uint4*)(o + ((size_t)b * EDIM + e0 + row) * NPIX + p0 + off) =
          *(const uint4*)(&Ob[row][off]);
    }
  } else {
    float* o = (float*)outv;
#pragma unroll
    for (int i = 0; i < 8; ++i) {
      int j = t + i * 128, row = j >> 4, off = (j & 15) * 4;
      *(uint4*)(o + ((size_t)b * EDIM + e0 + row) * NPIX + p0 + off) =
          *(const uint4*)(&Of[row][off]);
    }
  }
}

// ---------------------------------------------------------------------------
// Flash attention per (b, h, 64-query tile), transposed orientation.
// Q, K: (B, NPIX, EDIM) bf16 (Q pre-scaled by 1/8 via weights).
// V:    (B, EDIM, NPIX) bf16.  AO out: (B, NPIX, EDIM) bf16.
// Each lane owns ONE query (l16): softmax stats are lane-local scalars.
// ---------------------------------------------------------------------------
__global__ __launch_bounds__(128)
void attention_kernel(const __bf16* __restrict__ Q,
                      const __bf16* __restrict__ K,
                      const __bf16* __restrict__ V,
                      __bf16* __restrict__ AO) {
  __shared__ __bf16 qs[64][72];     // [query][d]; reused as output stage
  __shared__ __bf16 ks[64][72];     // [key][d]
  __shared__ __bf16 vst[64][72];    // [d][key]
  __shared__ __bf16 ps[4][16][72];  // per-wave P^T scratch: [query(l16)][key]

  const int t    = threadIdx.x;
  const int lane = t & 31;
  const int wave = t >> 5;
  const int half = lane >> 4;
  const int l16  = lane & 15;

  const int b  = blockIdx.z;
  const int h  = blockIdx.y;
  const int q0 = blockIdx.x * 64;

  const __bf16* Qb = Q + ((size_t)b * NPIX + q0) * EDIM + h * HDIM;
  const __bf16* Vb = V + ((size_t)b * EDIM + h * HDIM) * NPIX;

  // Stage Q tile (vectorized; 512 uint4, 4 per thread).
#pragma unroll
  for (int i = 0; i < 4; ++i) {
    int j = t + i * 128, row = j >> 3, off = (j & 7) * 8;
    *(uint4*)(&qs[row][off]) = *(const uint4*)(Qb + (size_t)row * EDIM + off);
  }
  __syncthreads();

  // Q^T B-fragments cached in registers for the entire key loop.
  v16bf bQ[2];
#pragma unroll
  for (int kstep = 0; kstep < 2; ++kstep)
    bQ[kstep] = load_frag_b(&qs[wave * 16 + l16][32 * kstep + 16 * half]);

  // acc[mt]: O^T fragment, M(d) = mt*16 + r + 8*half, N(query) = l16.
  v8f acc[4];
#pragma unroll
  for (int mt = 0; mt < 4; ++mt)
#pragma unroll
    for (int r = 0; r < 8; ++r) acc[mt][r] = 0.0f;

  float rm = -3.0e38f, rl = 0.0f;   // per-lane: stats for query l16

  for (int kt = 0; kt < NPIX; kt += 64) {
    const __bf16* Kb = K + ((size_t)b * NPIX + kt) * EDIM + h * HDIM;
    __syncthreads();
#pragma unroll
    for (int i = 0; i < 4; ++i) {
      int j = t + i * 128, row = j >> 3, off = (j & 7) * 8;
      *(uint4*)(&ks[row][off])  = *(const uint4*)(Kb + (size_t)row * EDIM + off);
      *(uint4*)(&vst[row][off]) = *(const uint4*)(Vb + (size_t)row * NPIX + kt + off);
      if (kt + 64 < NPIX) {
        __builtin_prefetch(Kb + 64 * EDIM + (size_t)row * EDIM + off, 0, 0);
        __builtin_prefetch(Vb + (size_t)row * NPIX + kt + 64 + off, 0, 0);
      }
    }
    __syncthreads();

    // S^T(64key x 16q) = K * Q^T: lane holds 32 keys of query l16.
    v8f sT[4];
#pragma unroll
    for (int mt = 0; mt < 4; ++mt)
#pragma unroll
      for (int r = 0; r < 8; ++r) sT[mt][r] = 0.0f;
#pragma unroll
    for (int kstep = 0; kstep < 2; ++kstep) {
      v16bf aK[4];
#pragma unroll
      for (int mt = 0; mt < 4; ++mt)
        aK[mt] = load_frag_a(&ks[mt * 16 + l16][32 * kstep], half);
#pragma unroll
      for (int mt = 0; mt < 4; ++mt)
        sT[mt] = __builtin_amdgcn_wmma_f32_16x16x32_bf16(
            false, aK[mt], false, bQ[kstep], (short)0, sT[mt], false, false);
    }

    // Online softmax: in-lane reduce over 32 keys + one xor-16 shuffle.
    float mx = -3.0e38f;
#pragma unroll
    for (int mt = 0; mt < 4; ++mt)
#pragma unroll
      for (int r = 0; r < 8; ++r) mx = fmaxf(mx, sT[mt][r]);
    mx = fmaxf(mx, __shfl_xor(mx, 16, 32));
    float mnew = fmaxf(rm, mx);
    float corr = __expf(rm - mnew);
    float rsum = 0.0f;
#pragma unroll
    for (int mt = 0; mt < 4; ++mt)
#pragma unroll
      for (int r = 0; r < 8; ++r) {
        float p = __expf(sT[mt][r] - mnew);
        sT[mt][r] = p;
        rsum += p;
      }
    rsum += __shfl_xor(rsum, 16, 32);
    rl = rl * corr + rsum;
    rm = mnew;
#pragma unroll
    for (int mt = 0; mt < 4; ++mt)
#pragma unroll
      for (int r = 0; r < 8; ++r) acc[mt][r] *= corr;

    // P^T -> ps[wave][query=l16][key]: 8 consecutive keys per (mt, half)
    // pack into one ds_store_b128 each (4 stores instead of 32).
#pragma unroll
    for (int mt = 0; mt < 4; ++mt)
      store_pack8_bf16(&ps[wave][l16][mt * 16 + 8 * half], sT[mt]);

    // O^T(64d x 16q) += V * P^T
#pragma unroll
    for (int kstep = 0; kstep < 2; ++kstep) {
      v16bf bP = load_frag_b(&ps[wave][l16][32 * kstep + 16 * half]);
      v16bf aV[4];
#pragma unroll
      for (int mt = 0; mt < 4; ++mt)
        aV[mt] = load_frag_a(&vst[mt * 16 + l16][32 * kstep], half);
#pragma unroll
      for (int mt = 0; mt < 4; ++mt)
        acc[mt] = __builtin_amdgcn_wmma_f32_16x16x32_bf16(
            false, aV[mt], false, bP, (short)0, acc[mt], false, false);
    }
  }

  // Epilogue: normalize (per-lane 1/rl), packed b128 stage into qs, then
  // coalesced global store.
  float inv = 1.0f / rl;
#pragma unroll
  for (int mt = 0; mt < 4; ++mt) {
    v8f v;
#pragma unroll
    for (int r = 0; r < 8; ++r) v[r] = acc[mt][r] * inv;
    store_pack8_bf16(&qs[wave * 16 + l16][mt * 16 + 8 * half], v);
  }
  __syncthreads();
#pragma unroll
  for (int i = 0; i < 4; ++i) {
    int j = t + i * 128, row = j >> 3, off = (j & 7) * 8;
    *(uint4*)(AO + ((size_t)b * NPIX + q0 + row) * EDIM + h * HDIM + off) =
        *(const uint4*)(&qs[row][off]);
  }
}

// ---------------------------------------------------------------------------
// Launch
// ---------------------------------------------------------------------------
extern "C" void kernel_launch(void* const* d_in, const int* in_sizes, int n_in,
                              void* d_out, int out_size, void* d_ws,
                              size_t ws_size, hipStream_t stream) {
  const float* x  = (const float*)d_in[0];
  const float* qw = (const float*)d_in[1];
  const float* qb = (const float*)d_in[2];
  const float* kw = (const float*)d_in[3];
  const float* kb = (const float*)d_in[4];
  const float* vw = (const float*)d_in[5];
  const float* vb = (const float*)d_in[6];
  const float* ow = (const float*)d_in[7];
  const float* ob = (const float*)d_in[8];
  float* out = (float*)d_out;

  char* ws = (char*)d_ws;
  size_t off = 0;
  __bf16* xt  = (__bf16*)(ws + off); off += (size_t)BDIM * NPIX * CDIM * 2;  // (B,N,C)
  __bf16* qwb = (__bf16*)(ws + off); off += (size_t)EDIM * CDIM * 2;
  __bf16* kwb = (__bf16*)(ws + off); off += (size_t)EDIM * CDIM * 2;
  __bf16* vwb = (__bf16*)(ws + off); off += (size_t)EDIM * CDIM * 2;
  __bf16* owb = (__bf16*)(ws + off); off += (size_t)EDIM * EDIM * 2;
  __bf16* Qm  = (__bf16*)(ws + off); off += (size_t)BDIM * NPIX * EDIM * 2;  // (B,N,E)
  __bf16* Km  = (__bf16*)(ws + off); off += (size_t)BDIM * NPIX * EDIM * 2;  // (B,N,E)
  __bf16* Vm  = (__bf16*)(ws + off); off += (size_t)BDIM * EDIM * NPIX * 2;  // (B,E,N)
  __bf16* AO  = (__bf16*)(ws + off); off += (size_t)BDIM * NPIX * EDIM * 2;  // (B,N,E)

  const float qscale = 0.125f;  // 1/sqrt(HD): folded into Q weights + bias

  transpose_cvt_kernel<<<dim3(NPIX / 32, CDIM / 32, BDIM), dim3(32, 8), 0, stream>>>(x, xt);
  auto cvt = [&](const float* src, __bf16* dst, int n, float s) {
    f32_to_bf16_kernel<<<(n + 255) / 256, 256, 0, stream>>>(src, dst, n, s);
  };
  cvt(qw, qwb, EDIM * CDIM, qscale);
  cvt(kw, kwb, EDIM * CDIM, 1.0f);
  cvt(vw, vwb, EDIM * CDIM, 1.0f);
  cvt(ow, owb, EDIM * EDIM, 1.0f);

  dim3 ggrid(NPIX / 64, EDIM / 64, BDIM);  // (16, 8, 8)
  gemm_bias_kernel<0><<<ggrid, 128, 0, stream>>>(xt, qwb, qb, Qm, CDIM, qscale);
  gemm_bias_kernel<0><<<ggrid, 128, 0, stream>>>(xt, kwb, kb, Km, CDIM, 1.0f);
  gemm_bias_kernel<1><<<ggrid, 128, 0, stream>>>(xt, vwb, vb, Vm, CDIM, 1.0f);

  dim3 agrid(NPIX / 64, NHEAD, BDIM);      // (16, 8, 8)
  attention_kernel<<<agrid, 128, 0, stream>>>(Qm, Km, Vm, AO);

  gemm_bias_kernel<2><<<ggrid, 128, 0, stream>>>(AO, owb, ob, out, EDIM, 1.0f);
}
